// DecoderNet_40338332844541
// MI455X (gfx1250) — matrix operand order repaired and checked by reading, not compiled, not run on hardware
//
#include <hip/hip_runtime.h>
#include <hip/hip_bf16.h>
#include <math.h>

typedef __attribute__((ext_vector_type(16))) _Float16 v16h;
typedef __attribute__((ext_vector_type(8)))  _Float16 v8h;
typedef __attribute__((ext_vector_type(8)))  float    v8f;

#define HIDDEN 512
#define WORD   1024
#define VOCAB  4004
#define BATCH  512
#define SEQLEN 80
#define TSTEPS 27
#define BS     (BATCH * SEQLEN)   // 40960
#define G4H    (4 * HIDDEN)       // 2048

#define LD8(p) (*(const v8h*)(p))

// ---------------------------------------------------------------------------
// WMMA GEMM:  Y[M,N] = X[M,K] @ W[N,K]^T (+ bias[N])
// X row-major f16 (ldX), W row-major f16 [out,in] (ldW).
// Each wave computes a 32x32 tile as 2x2 WMMA 16x16 tiles (A/B frags reused
// 2x each => 8 b128 loads per 4 WMMAs). Block = 8 waves = 64(M) x 128(N).
// OOB rows/cols are CLAMPED on load (garbage only lands in never-stored
// rows/cols of D) so the inner loop has zero divergence; stores are guarded.
// 2-stage software pipeline overlaps next-K loads with current WMMAs.
// ---------------------------------------------------------------------------
__global__ __launch_bounds__(256)
void gemm_wmma_f16(const _Float16* __restrict__ X, const _Float16* __restrict__ W,
                   const float* __restrict__ bias,
                   float* __restrict__ Yf, _Float16* __restrict__ Yh,
                   int M, int N, int K, int ldX, int ldW, int ldY)
{
    const int lane = threadIdx.x & 31;
    const int wid  = threadIdx.x >> 5;
    const int half = lane >> 4;
    const int r    = lane & 15;

    const int m0 = blockIdx.x * 64  + (wid & 1) * 32;
    const int n0 = blockIdx.y * 128 + (wid >> 1) * 32;

    // clamped per-lane source rows/cols (loads always in-bounds)
    const int ar0 = min(m0 + r,      M - 1);
    const int ar1 = min(m0 + 16 + r, M - 1);
    const int bc0 = min(n0 + r,      N - 1);
    const int bc1 = min(n0 + 16 + r, N - 1);

    const _Float16* apA = X + (size_t)ar0 * ldX + half * 8;
    const _Float16* apB = X + (size_t)ar1 * ldX + half * 8;
    const _Float16* bpA = W + (size_t)bc0 * ldW + half * 8;
    const _Float16* bpB = W + (size_t)bc1 * ldW + half * 8;

    v8f c00 = {}, c01 = {}, c10 = {}, c11 = {};

    // pipeline stage registers
    v8h a0l = LD8(apA),      a0h = LD8(apA + 16);
    v8h a1l = LD8(apB),      a1h = LD8(apB + 16);
    v8h b0l = LD8(bpA),      b0h = LD8(bpA + 16);
    v8h b1l = LD8(bpB),      b1h = LD8(bpB + 16);

    auto compute = [&]() {
        v16h A0, A1, B0, B1;
#pragma unroll
        for (int i = 0; i < 8; ++i) {
            A0[i] = a0l[i]; A0[i + 8] = a0h[i];
            A1[i] = a1l[i]; A1[i + 8] = a1h[i];
            B0[i] = b0l[i]; B0[i + 8] = b0h[i];
            B1[i] = b1l[i]; B1[i + 8] = b1h[i];
        }
        c00 = __builtin_amdgcn_wmma_f32_16x16x32_f16(false, A0, false, B0, (short)0, c00, false, false);
        c01 = __builtin_amdgcn_wmma_f32_16x16x32_f16(false, A0, false, B1, (short)0, c01, false, false);
        c10 = __builtin_amdgcn_wmma_f32_16x16x32_f16(false, A1, false, B0, (short)0, c10, false, false);
        c11 = __builtin_amdgcn_wmma_f32_16x16x32_f16(false, A1, false, B1, (short)0, c11, false, false);
    };

    for (int k0 = 32; k0 < K; k0 += 32) {
        // issue next-stage loads before computing on current stage
        v8h na0l = LD8(apA + k0), na0h = LD8(apA + k0 + 16);
        v8h na1l = LD8(apB + k0), na1h = LD8(apB + k0 + 16);
        v8h nb0l = LD8(bpA + k0), nb0h = LD8(bpA + k0 + 16);
        v8h nb1l = LD8(bpB + k0), nb1h = LD8(bpB + k0 + 16);
        compute();
        a0l = na0l; a0h = na0h; a1l = na1l; a1h = na1h;
        b0l = nb0l; b0h = nb0h; b1l = nb1l; b1h = nb1h;
    }
    compute();

    // ---- store 2x2 tiles with guards ----
    const int col0 = n0 + r, col1 = n0 + 16 + r;
    const float bv0 = (bias && col0 < N) ? bias[col0] : 0.0f;
    const float bv1 = (bias && col1 < N) ? bias[col1] : 0.0f;
#pragma unroll
    for (int j = 0; j < 8; ++j) {
        const int row0 = m0 + half * 8 + j;
        const int row1 = row0 + 16;
        if (row0 < M) {
            if (col0 < N) {
                const float v = c00[j] + bv0;
                if (Yf) Yf[(size_t)row0 * ldY + col0] = v;
                if (Yh) Yh[(size_t)row0 * ldY + col0] = (_Float16)v;
            }
            if (col1 < N) {
                const float v = c01[j] + bv1;
                if (Yf) Yf[(size_t)row0 * ldY + col1] = v;
                if (Yh) Yh[(size_t)row0 * ldY + col1] = (_Float16)v;
            }
        }
        if (row1 < M) {
            if (col0 < N) {
                const float v = c10[j] + bv0;
                if (Yf) Yf[(size_t)row1 * ldY + col0] = v;
                if (Yh) Yh[(size_t)row1 * ldY + col0] = (_Float16)v;
            }
            if (col1 < N) {
                const float v = c11[j] + bv1;
                if (Yf) Yf[(size_t)row1 * ldY + col1] = v;
                if (Yh) Yh[(size_t)row1 * ldY + col1] = (_Float16)v;
            }
        }
    }
}

// --------------------------- support kernels -------------------------------
__global__ void cvt_f16(const float* __restrict__ src, _Float16* __restrict__ dst, int n)
{
    int i = blockIdx.x * blockDim.x + threadIdx.x;
    if (i < n) dst[i] = (_Float16)src[i];
}

// slice columns [col0, col0+cols) out of src[rows, ldsrc] -> dst[rows, cols] f16
__global__ void slice_f16(const float* __restrict__ src, _Float16* __restrict__ dst,
                          int rows, int cols, int ldsrc, int col0)
{
    int i = blockIdx.x * blockDim.x + threadIdx.x;
    if (i >= rows * cols) return;
    int rr = i / cols, cc = i - rr * cols;
    dst[i] = (_Float16)src[(size_t)rr * ldsrc + col0 + cc];
}

__global__ void vec_add(const float* __restrict__ a, const float* __restrict__ b,
                        float* __restrict__ o, int n)
{
    int i = blockIdx.x * blockDim.x + threadIdx.x;
    if (i < n) o[i] = a[i] + b[i];
}

// scores[row] = dot(X[row,:], w)  -- one wave per row
__global__ void dot_rows(const _Float16* __restrict__ X, const _Float16* __restrict__ w,
                         float* __restrict__ out, int rows, int K)
{
    int row  = blockIdx.x * (blockDim.x >> 5) + (threadIdx.x >> 5);
    int lane = threadIdx.x & 31;
    if (row >= rows) return;
    const _Float16* p = X + (size_t)row * K;
    float s = 0.f;
    for (int k = lane; k < K; k += 32) s += (float)p[k] * (float)w[k];
#pragma unroll
    for (int off = 16; off > 0; off >>= 1) s += __shfl_down(s, off, 32);
    if (lane == 0) out[row] = s;
}

__global__ void softmax_seq(float* __restrict__ scores)
{
    __shared__ float buf[SEQLEN];
    __shared__ float mx, sm;
    float* p = scores + (size_t)blockIdx.x * SEQLEN;
    int t = threadIdx.x;
    if (t < SEQLEN) buf[t] = p[t];
    __syncthreads();
    if (t == 0) { float m = buf[0]; for (int s = 1; s < SEQLEN; ++s) m = fmaxf(m, buf[s]); mx = m; }
    __syncthreads();
    if (t < SEQLEN) buf[t] = expf(buf[t] - mx);
    __syncthreads();
    if (t == 0) { float s2 = 0.f; for (int s = 0; s < SEQLEN; ++s) s2 += buf[s]; sm = s2; }
    __syncthreads();
    if (t < SEQLEN) p[t] = buf[t] / sm;
}

// ctx[b,h] = sum_s attw[b,s] * enc[b,s,h]   (one block per b, 512 threads)
__global__ void context_kernel(const float* __restrict__ attw, const float* __restrict__ enc,
                               float* __restrict__ ctx, _Float16* __restrict__ ctxh)
{
    int b = blockIdx.x, h = threadIdx.x;
    const float* aw = attw + (size_t)b * SEQLEN;
    const float* e  = enc + (size_t)b * SEQLEN * HIDDEN + h;
    float s = 0.f;
    for (int q = 0; q < SEQLEN; ++q) s = fmaf(aw[q], e[(size_t)q * HIDDEN], s);
    ctx[(size_t)b * HIDDEN + h]  = s;
    ctxh[(size_t)b * HIDDEN + h] = (_Float16)s;
}

__global__ void init_state(const float* __restrict__ h0, float* __restrict__ h,
                           _Float16* __restrict__ hh, float* __restrict__ c,
                           int* __restrict__ tok)
{
    int i = blockIdx.x * blockDim.x + threadIdx.x;
    if (i < BATCH * HIDDEN) {
        float v = h0[i];
        h[i] = v; hh[i] = (_Float16)v; c[i] = 0.f;
    }
    if (i < BATCH) tok[i] = 1;
}

__device__ __forceinline__ float sigm(float x) { return 1.f / (1.f + expf(-x)); }

__global__ void lstm_cell(const float* __restrict__ TG, const float* __restrict__ gctx,
                          const float* __restrict__ gh, const int* __restrict__ tok,
                          float* __restrict__ c, float* __restrict__ h,
                          _Float16* __restrict__ hf)
{
    int i = blockIdx.x * blockDim.x + threadIdx.x;
    if (i >= BATCH * HIDDEN) return;
    int b = i >> 9, j = i & (HIDDEN - 1);
    size_t gb = (size_t)b * G4H;
    size_t tb = (size_t)tok[b] * G4H;
    float gi = TG[tb + j]          + gctx[gb + j]          + gh[gb + j];
    float gf = TG[tb + 512 + j]    + gctx[gb + 512 + j]    + gh[gb + 512 + j];
    float gg = TG[tb + 1024 + j]   + gctx[gb + 1024 + j]   + gh[gb + 1024 + j];
    float go = TG[tb + 1536 + j]   + gctx[gb + 1536 + j]   + gh[gb + 1536 + j];
    float cn = sigm(gf) * c[i] + sigm(gi) * tanhf(gg);
    float hn = sigm(go) * tanhf(cn);
    c[i] = cn; h[i] = hn; hf[i] = (_Float16)hn;
}

// argmax over VOCAB for each batch row; writes next token + float prediction
__global__ void argmax_vocab(const float* __restrict__ logits, int ldY,
                             int* __restrict__ tok, float* __restrict__ pred, int t)
{
    __shared__ float smax[256];
    __shared__ int   sidx[256];
    int b = blockIdx.x, tt = threadIdx.x;
    const float* p = logits + (size_t)b * ldY;
    float mv = -3.4e38f; int mi = 0;
    for (int v = tt; v < VOCAB; v += 256) {
        float x = p[v];
        if (x > mv) { mv = x; mi = v; }
    }
    smax[tt] = mv; sidx[tt] = mi;
    __syncthreads();
    for (int off = 128; off > 0; off >>= 1) {
        if (tt < off) {
            bool take = (smax[tt + off] > smax[tt]) ||
                        (smax[tt + off] == smax[tt] && sidx[tt + off] < sidx[tt]);
            if (take) { smax[tt] = smax[tt + off]; sidx[tt] = sidx[tt + off]; }
        }
        __syncthreads();
    }
    if (tt == 0) {
        tok[b] = sidx[0];
        pred[(size_t)b * TSTEPS + t] = (float)sidx[0];
    }
}

// ---------------------------------------------------------------------------
extern "C" void kernel_launch(void* const* d_in, const int* in_sizes, int n_in,
                              void* d_out, int out_size, void* d_ws, size_t ws_size,
                              hipStream_t stream)
{
    (void)in_sizes; (void)n_in; (void)out_size; (void)ws_size;
    const float* enc_last = (const float*)d_in[0];
    const float* enc      = (const float*)d_in[1];
    const float* emb      = (const float*)d_in[2];
    const float* w1       = (const float*)d_in[3];
    const float* b1       = (const float*)d_in[4];
    const float* w2       = (const float*)d_in[5];
    const float* b2       = (const float*)d_in[6];
    const float* w3       = (const float*)d_in[7];
    const float* b3       = (const float*)d_in[8];
    const float* wv       = (const float*)d_in[9];
    const float* w_ih     = (const float*)d_in[10];
    const float* w_hh     = (const float*)d_in[11];
    const float* b_ih     = (const float*)d_in[12];
    const float* b_hh     = (const float*)d_in[13];
    const float* out_w    = (const float*)d_in[14];
    const float* out_b    = (const float*)d_in[15];

    float* seq  = (float*)d_out;                                  // [B, T, VOCAB]
    float* pred = seq + (size_t)BATCH * TSTEPS * VOCAB;           // [B, T] (as float)

    // ---- workspace bump allocator ----
    char* ws = (char*)d_ws;
    size_t off = 0;
    auto alloc = [&](size_t bytes) -> void* {
        void* p = ws + off;
        off = (off + bytes + 255) & ~(size_t)255;
        return p;
    };
    _Float16* enc_h  = (_Float16*)alloc((size_t)BS * HIDDEN * 2);   // also reused for e3
    _Float16* e1h    = (_Float16*)alloc((size_t)BS * HIDDEN * 2);
    _Float16* e2h    = (_Float16*)alloc((size_t)BS * HIDDEN * 2);
    _Float16* w1e_h  = (_Float16*)alloc((size_t)HIDDEN * HIDDEN * 2);
    _Float16* w2_h   = (_Float16*)alloc((size_t)HIDDEN * HIDDEN * 2);
    _Float16* w3_h   = (_Float16*)alloc((size_t)HIDDEN * HIDDEN * 2);
    _Float16* wv_h   = (_Float16*)alloc((size_t)HIDDEN * 2);
    _Float16* emb_h  = (_Float16*)alloc((size_t)VOCAB * WORD * 2);
    _Float16* outw_h = (_Float16*)alloc((size_t)VOCAB * HIDDEN * 2);
    _Float16* whh_h  = (_Float16*)alloc((size_t)G4H * HIDDEN * 2);
    _Float16* wihw_h = (_Float16*)alloc((size_t)G4H * WORD * 2);
    _Float16* wihc_h = (_Float16*)alloc((size_t)G4H * HIDDEN * 2);
    float*    bsum   = (float*)alloc((size_t)G4H * 4);
    float*    scores = (float*)alloc((size_t)BS * 4);             // becomes attw in place
    float*    ctx    = (float*)alloc((size_t)BATCH * HIDDEN * 4);
    _Float16* ctx_h  = (_Float16*)alloc((size_t)BATCH * HIDDEN * 2);
    float*    gctx   = (float*)alloc((size_t)BATCH * G4H * 4);
    float*    TG     = (float*)alloc((size_t)VOCAB * G4H * 4);
    float*    gh     = (float*)alloc((size_t)BATCH * G4H * 4);
    float*    hbuf   = (float*)alloc((size_t)BATCH * HIDDEN * 4);
    float*    cbuf   = (float*)alloc((size_t)BATCH * HIDDEN * 4);
    _Float16* h_h    = (_Float16*)alloc((size_t)BATCH * HIDDEN * 2);
    int*      tok    = (int*)alloc((size_t)BATCH * 4);

    auto cdiv = [](int a, int b) { return (a + b - 1) / b; };
    const dim3 blk(256);

    // ---- one-time precompute ----
    cvt_f16<<<cdiv(BS * HIDDEN, 256), blk, 0, stream>>>(enc, enc_h, BS * HIDDEN);
    slice_f16<<<cdiv(HIDDEN * HIDDEN, 256), blk, 0, stream>>>(w1, w1e_h, HIDDEN, HIDDEN, 2 * HIDDEN, 0);
    cvt_f16<<<cdiv(HIDDEN * HIDDEN, 256), blk, 0, stream>>>(w2, w2_h, HIDDEN * HIDDEN);
    cvt_f16<<<cdiv(HIDDEN * HIDDEN, 256), blk, 0, stream>>>(w3, w3_h, HIDDEN * HIDDEN);
    cvt_f16<<<cdiv(HIDDEN, 256), blk, 0, stream>>>(wv, wv_h, HIDDEN);
    cvt_f16<<<cdiv(VOCAB * WORD, 256), blk, 0, stream>>>(emb, emb_h, VOCAB * WORD);
    cvt_f16<<<cdiv(VOCAB * HIDDEN, 256), blk, 0, stream>>>(out_w, outw_h, VOCAB * HIDDEN);
    cvt_f16<<<cdiv(G4H * HIDDEN, 256), blk, 0, stream>>>(w_hh, whh_h, G4H * HIDDEN);
    slice_f16<<<cdiv(G4H * WORD, 256), blk, 0, stream>>>(w_ih, wihw_h, G4H, WORD, WORD + HIDDEN, 0);
    slice_f16<<<cdiv(G4H * HIDDEN, 256), blk, 0, stream>>>(w_ih, wihc_h, G4H, HIDDEN, WORD + HIDDEN, WORD);
    vec_add<<<cdiv(G4H, 256), blk, 0, stream>>>(b_ih, b_hh, bsum, G4H);

    // attention chain over the full [B*S] rows (constant across time steps)
    gemm_wmma_f16<<<dim3(cdiv(BS, 64), cdiv(HIDDEN, 128)), blk, 0, stream>>>(
        enc_h, w1e_h, b1, nullptr, e1h, BS, HIDDEN, HIDDEN, HIDDEN, HIDDEN, HIDDEN);
    gemm_wmma_f16<<<dim3(cdiv(BS, 64), cdiv(HIDDEN, 128)), blk, 0, stream>>>(
        e1h, w2_h, b2, nullptr, e2h, BS, HIDDEN, HIDDEN, HIDDEN, HIDDEN, HIDDEN);
    gemm_wmma_f16<<<dim3(cdiv(BS, 64), cdiv(HIDDEN, 128)), blk, 0, stream>>>(
        e2h, w3_h, b3, nullptr, enc_h /*reuse as e3*/, BS, HIDDEN, HIDDEN, HIDDEN, HIDDEN, HIDDEN);
    dot_rows<<<cdiv(BS, 8), blk, 0, stream>>>(enc_h, wv_h, scores, BS, HIDDEN);
    softmax_seq<<<BATCH, 128, 0, stream>>>(scores);
    context_kernel<<<BATCH, HIDDEN, 0, stream>>>(scores, enc, ctx, ctx_h);

    // constant gate contributions
    gemm_wmma_f16<<<dim3(cdiv(BATCH, 64), cdiv(G4H, 128)), blk, 0, stream>>>(
        ctx_h, wihc_h, bsum, gctx, nullptr, BATCH, G4H, HIDDEN, HIDDEN, HIDDEN, G4H);
    gemm_wmma_f16<<<dim3(cdiv(VOCAB, 64), cdiv(G4H, 128)), blk, 0, stream>>>(
        emb_h, wihw_h, nullptr, TG, nullptr, VOCAB, G4H, WORD, WORD, WORD, G4H);

    init_state<<<cdiv(BATCH * HIDDEN, 256), blk, 0, stream>>>(enc_last, hbuf, h_h, cbuf, tok);

    // ---- 27 sequential decode steps ----
    for (int t = 0; t < TSTEPS; ++t) {
        gemm_wmma_f16<<<dim3(cdiv(BATCH, 64), cdiv(G4H, 128)), blk, 0, stream>>>(
            h_h, whh_h, nullptr, gh, nullptr, BATCH, G4H, HIDDEN, HIDDEN, HIDDEN, G4H);
        lstm_cell<<<cdiv(BATCH * HIDDEN, 256), blk, 0, stream>>>(
            TG, gctx, gh, tok, cbuf, hbuf, h_h);
        gemm_wmma_f16<<<dim3(cdiv(BATCH, 64), cdiv(VOCAB, 128)), blk, 0, stream>>>(
            h_h, outw_h, out_b, seq + (size_t)t * VOCAB, nullptr,
            BATCH, VOCAB, HIDDEN, HIDDEN, HIDDEN, TSTEPS * VOCAB);
        argmax_vocab<<<BATCH, blk, 0, stream>>>(
            seq + (size_t)t * VOCAB, TSTEPS * VOCAB, tok, pred, t);
    }
}
